// DDSP_23184233464200
// MI455X (gfx1250) — matrix operand order repaired and checked by reading, not compiled, not run on hardware
//
#include <hip/hip_runtime.h>
#include <hip/hip_bf16.h>
#include <math.h>

typedef _Float16 f16;
typedef __attribute__((ext_vector_type(16))) _Float16 v16h;
typedef __attribute__((ext_vector_type(8)))  float    v8f;
typedef int v4i_ __attribute__((vector_size(16)));   // matches async-LDS builtin param type

#define HID    512
#define LAT    16
#define NHARM  64
#define NBANDS 65
#define SRATE  16000
#define BLOCK  160
#define BB     8
#define TT     400
#define ROWS   (BB*TT)        // 3200
#define SIGLEN (TT*BLOCK)     // 64000
#define TWO_PI 6.28318530717958647692f

#if __has_builtin(__builtin_amdgcn_global_load_async_to_lds_b128) && \
    __has_builtin(__builtin_amdgcn_s_wait_asynccnt)
#define HAVE_ASYNC_LDS 1
#else
#define HAVE_ASYNC_LDS 0
#endif

// ---------------------------------------------------------------------------
// WMMA fragment helpers (CDNA5 wave32 layouts, CDNA5 ISA 7.12.2)
// ---------------------------------------------------------------------------
__device__ __forceinline__ v16h load_a_frag_lds(const f16* As /*16x32*/) {
  int lane = threadIdx.x & 31;
  int m = lane & 15, half = lane >> 4;
  v16h a;
#pragma unroll
  for (int v = 0; v < 8; ++v) {
    int k0 = (v >> 2) * 16 + half * 8 + (v & 3) * 2;
    a[2 * v]     = As[m * 32 + k0];
    a[2 * v + 1] = As[m * 32 + k0 + 1];
  }
  return a;
}

__device__ __forceinline__ v16h load_b_frag_lds(const f16* Ws /*16x32 tile*/) {
  int lane = threadIdx.x & 31;
  int n = lane & 15, half = lane >> 4;
  v16h b;
#pragma unroll
  for (int v = 0; v < 8; ++v) {
    int k0 = (v >> 2) * 16 + half * 8 + (v & 3) * 2;
    b[2 * v]     = Ws[n * 32 + k0];
    b[2 * v + 1] = Ws[n * 32 + k0 + 1];
  }
  return b;
}

// ---------------------------------------------------------------------------
// GEMM: C(MxN) = A(MxK,f32) * W^T (W is (N,K) f16 row-major) + bias.
// 256 threads (8 waves); block tile 64(M) x 128(N); each wave owns one
// 16-wide n-subtile and 4 m-subtiles -> 4 wmma per K-chunk, B-frag reused 4x.
// Weight tiles staged LDS-direct via GLOBAL_LOAD_ASYNC_TO_LDS_B128 when K%32==0.
// A tiles staged branch-free (float2 -> packed 2xf16) on the fast path.
// ---------------------------------------------------------------------------
__global__ __launch_bounds__(256) void gemm_bias_wmma(
    const float* __restrict__ A, int lda,
    const f16* __restrict__ W,
    const float* __restrict__ bias,
    float* __restrict__ C, int ldc,
    int M, int N, int K) {
  __shared__ alignas(16) f16 As[64 * 32];    // 4 KB
  __shared__ alignas(16) f16 Ws[128 * 32];   // 8 KB
  int tid = threadIdx.x;
  int wave = tid >> 5, lane = tid & 31;
  int m0 = blockIdx.y * 64;
  int nblk = blockIdx.x * 128;
  // fast path: whole 32-chunks valid, full 64-row M tile, float2-aligned rows
  const bool fastA = ((K & 31) == 0) && (m0 + 64 <= M) && ((lda & 1) == 0);
  const bool asyncW = ((K & 31) == 0);
  v8f acc[4] = {};
  for (int kc = 0; kc < K; kc += 32) {
    // ---- stage A tile 64x32 (f32 -> f16) ----
    if (fastA) {
      // 1024 float2 chunks, 4 per thread, no guards, packed 32-bit LDS stores
#pragma unroll
      for (int it = 0; it < 4; ++it) {
        int i = tid + it * 256;
        int m = i >> 4, k = (i & 15) * 2;
        const float* src = &A[(size_t)(m0 + m) * lda + kc + k];
        float x0 = src[0], x1 = src[1];
        union { f16 h[2]; unsigned u; } pk;
        pk.h[0] = (f16)x0; pk.h[1] = (f16)x1;
        *(unsigned*)&As[m * 32 + k] = pk.u;
      }
    } else {
      for (int i = tid; i < 64 * 32; i += 256) {
        int m = i >> 5, k = i & 31;
        int gm = m0 + m, gk = kc + k;
        As[i] = (gm < M && gk < K) ? (f16)A[(size_t)gm * lda + gk] : (f16)0.f;
      }
    }
    // ---- stage W tile 128x32 (f16) ----
#if HAVE_ASYNC_LDS
    if (asyncW) {
      // 128 rows x 32 f16 = 512 x 16B chunks; async DMA straight into LDS
      for (int c = tid; c < 512; c += 256) {
        int n = c >> 2, koff = (c & 3) * 8;
        int gn = nblk + n;
        f16* dst = &Ws[n * 32 + koff];
        if (gn < N) {
          __builtin_amdgcn_global_load_async_to_lds_b128(
              (__attribute__((address_space(1))) v4i_*)(W + (size_t)gn * K + kc + koff),
              (__attribute__((address_space(3))) v4i_*)dst, 0, 0);
        } else {
#pragma unroll
          for (int q = 0; q < 8; ++q) dst[q] = (f16)0.f;
        }
      }
      __builtin_amdgcn_s_wait_asynccnt(0);
    } else
#endif
    {
      for (int i = tid; i < 128 * 32; i += 256) {
        int n = i >> 5, k = i & 31;
        int gn = nblk + n, gk = kc + k;
        Ws[i] = (gn < N && gk < K) ? W[(size_t)gn * K + gk] : (f16)0.f;
      }
      if (kc + 32 < K) {
        int gn = nblk + (tid >> 1);
        if (gn < N) __builtin_prefetch(&W[(size_t)gn * K + kc + 32], 0, 1);
      }
    }
    __syncthreads();
    v16h b = load_b_frag_lds(&Ws[(wave * 16) * 32]);
#pragma unroll
    for (int mi = 0; mi < 4; ++mi) {
      v16h a = load_a_frag_lds(&As[(mi * 16) * 32]);
      acc[mi] = __builtin_amdgcn_wmma_f32_16x16x32_f16(false, a, false, b,
                                                       (short)0, acc[mi], false, false);
    }
    __syncthreads();
  }
  int n = nblk + wave * 16 + (lane & 15);
  int half = lane >> 4;
  if (n < N) {
    float bv = bias ? bias[n] : 0.f;
#pragma unroll
    for (int mi = 0; mi < 4; ++mi) {
#pragma unroll
      for (int r = 0; r < 8; ++r) {
        int m = m0 + mi * 16 + half * 8 + r;
        if (m < M) C[(size_t)m * ldc + n] = acc[mi][r] + bv;
      }
    }
  }
}

// ---------------------------------------------------------------------------
// Persistent sequential GRU: one 1024-thread workgroup (32 waves), H=512.
// gates(8x1536) via WMMA from LDS-resident h; W_hh f16 streamed from L2.
// ---------------------------------------------------------------------------
__global__ __launch_bounds__(1024) void gru_seq(
    const float* __restrict__ xw,     // (B,T,1536)
    const f16*  __restrict__ Whh,     // (1536,512) f16
    const float* __restrict__ bhh,    // (1536)
    float* __restrict__ hs,           // (B,T,512)
    int T) {
  __shared__ float hsh[BB * HID];     // 16 KB
  __shared__ f16   gsh[BB * 1536];    // 24 KB
  int tid = threadIdx.x;
  int wave = tid >> 5, lane = tid & 31;
  for (int i = tid; i < BB * HID; i += 1024) hsh[i] = 0.f;
  __syncthreads();
  for (int t = 0; t < T; ++t) {
#pragma unroll
    for (int j = 0; j < 3; ++j) {
      int n0 = (wave * 3 + j) << 4;      // 96 n-tiles over 32 waves
      v8f acc = {};
      for (int k0 = 0; k0 < HID; k0 += 32) {
        int m = lane & 15, half = lane >> 4;
        v16h a;
#pragma unroll
        for (int v = 0; v < 8; ++v) {
          int kk = k0 + (v >> 2) * 16 + half * 8 + (v & 3) * 2;
          if (m < BB) {
            a[2 * v]     = (f16)hsh[m * HID + kk];
            a[2 * v + 1] = (f16)hsh[m * HID + kk + 1];
          } else {
            a[2 * v] = (f16)0.f; a[2 * v + 1] = (f16)0.f;
          }
        }
        int n = n0 + (lane & 15);
        v16h b;
#pragma unroll
        for (int v = 0; v < 8; ++v) {
          int kk = k0 + (v >> 2) * 16 + half * 8 + (v & 3) * 2;
          b[2 * v]     = Whh[(size_t)n * HID + kk];
          b[2 * v + 1] = Whh[(size_t)n * HID + kk + 1];
        }
        acc = __builtin_amdgcn_wmma_f32_16x16x32_f16(false, a, false, b,
                                                     (short)0, acc, false, false);
      }
      int n = n0 + (lane & 15), half = lane >> 4;
#pragma unroll
      for (int r = 0; r < 8; ++r) {
        int m = half * 8 + r;
        if (m < BB) gsh[m * 1536 + n] = (f16)acc[r];
      }
    }
    __syncthreads();
    for (int i = tid; i < BB * HID; i += 1024) {
      int b = i >> 9, j = i & (HID - 1);
      const float* gi = &xw[((size_t)b * T + t) * 1536];
      float ghr = (float)gsh[b * 1536 + j]            + bhh[j];
      float ghz = (float)gsh[b * 1536 + HID + j]      + bhh[HID + j];
      float ghn = (float)gsh[b * 1536 + 2 * HID + j]  + bhh[2 * HID + j];
      float r = 1.f / (1.f + __expf(-(gi[j] + ghr)));
      float z = 1.f / (1.f + __expf(-(gi[HID + j] + ghz)));
      float nn = tanhf(gi[2 * HID + j] + r * ghn);
      float hh = (1.f - z) * nn + z * hsh[i];
      hsh[i] = hh;
      hs[((size_t)b * T + t) * HID + j] = hh;
    }
    __syncthreads();
  }
}

// ---------------------------------------------------------------------------
// LayerNorm (+ optional leaky-relu), one block per row.
// ---------------------------------------------------------------------------
__global__ __launch_bounds__(256) void ln_act(
    const float* __restrict__ src, int lds,
    float* __restrict__ dst, int ldd,
    const float* __restrict__ g, const float* __restrict__ bb,
    int width, int leaky) {
  __shared__ float rs[256], rq[256];
  int row = blockIdx.x, tid = threadIdx.x;
  const float* x = src + (size_t)row * lds;
  float s = 0.f, s2 = 0.f;
  for (int i = tid; i < width; i += 256) { float v = x[i]; s += v; s2 += v * v; }
  rs[tid] = s; rq[tid] = s2; __syncthreads();
  for (int off = 128; off > 0; off >>= 1) {
    if (tid < off) { rs[tid] += rs[tid + off]; rq[tid] += rq[tid + off]; }
    __syncthreads();
  }
  float mean = rs[0] / width;
  float var  = rq[0] / width - mean * mean;
  float inv  = rsqrtf(var + 1e-5f);
  for (int i = tid; i < width; i += 256) {
    float v = (x[i] - mean) * inv * g[i] + bb[i];
    if (leaky) v = v >= 0.f ? v : 0.01f * v;
    dst[(size_t)row * ldd + i] = v;
  }
}

__global__ void cvt_f16_kernel(const float* __restrict__ s, f16* __restrict__ d, int n) {
  int i = blockIdx.x * blockDim.x + threadIdx.x;
  if (i < n) d[i] = (f16)s[i];
}

__global__ void z_kernel(const float* mu, const float* lv, const float* eps,
                         float* z, int n) {
  int i = blockIdx.x * blockDim.x + threadIdx.x;
  if (i < n) z[i] = mu[i] + eps[i] * __expf(0.5f * lv[i]);
}

__global__ void concat_hidden2(const float* hs, const float* pitch,
                               const float* loud, float* out, int n) {
  int i = blockIdx.x * blockDim.x + threadIdx.x;
  if (i >= n) return;
  int r = i / 514, c = i - r * 514;
  out[i] = (c < HID) ? hs[(size_t)r * HID + c] : (c == HID ? pitch[r] : loud[r]);
}

__device__ __forceinline__ float scale_fn(float x) {
  float sg = 1.f / (1.f + __expf(-x));
  return 2.f * __powf(sg, 2.302585092994046f) + 1e-7f;
}

// p = scale_fn(proj0); mask by nyquist; normalize; scale by total_amp -> amps(rows,64)
__global__ __launch_bounds__(64) void amps_kernel(const float* __restrict__ plin,
                                                  const float* __restrict__ pitch,
                                                  float* __restrict__ amps) {
  __shared__ float red[64];
  int row = blockIdx.x, t = threadIdx.x;
  float p  = scale_fn(plin[(size_t)row * NBANDS + 1 + t]);
  float pt = pitch[row];
  float aa = ((pt * (float)(t + 1)) < (0.5f * SRATE) ? 1.f : 0.f) + 1e-4f;
  p *= aa;
  red[t] = p; __syncthreads();
  for (int off = 32; off > 0; off >>= 1) {
    if (t < off) red[t] += red[t + off];
    __syncthreads();
  }
  float total = scale_fn(plin[(size_t)row * NBANDS]);
  amps[(size_t)row * NHARM + t] = p / red[0] * total;
}

// bands = scale_fn(proj1 - 5); amp_to_ir (irfft(65) -> roll/hann/pad/roll) -> (rows,160)
__global__ __launch_bounds__(160) void impulse_kernel(const float* __restrict__ blin,
                                                      float* __restrict__ impulse) {
  __shared__ float bands[NBANDS];
  int row = blockIdx.x, i = threadIdx.x;
  for (int k = i; k < NBANDS; k += 160)
    bands[k] = scale_fn(blin[(size_t)row * NBANDS + k] - 5.f);
  __syncthreads();
  int j = (i + 64) % 160;         // undo final roll(-64) over len 160
  float val = 0.f;
  if (j < 128) {
    int src = (j + 64) & 127;     // undo roll(+64) over len 128
    float acc = bands[0];
    for (int k = 1; k < 64; ++k)
      acc += 2.f * bands[k] * __cosf(TWO_PI * (float)(k * src) / 128.f);
    acc += bands[64] * __cosf(3.14159265358979f * (float)src);
    acc *= (1.f / 128.f);
    float win = 0.5f * (1.f - __cosf(TWO_PI * (float)j / 128.f));
    val = acc * win;
  }
  impulse[(size_t)row * BLOCK + i] = val;
}

// per-frame causal 160-tap conv of (2*noise_u-1) with impulse
__global__ __launch_bounds__(160) void noise_conv(const float* __restrict__ nu,
                                                  const float* __restrict__ impulse,
                                                  float* __restrict__ out) {
  __shared__ float s[BLOCK], k[BLOCK];
  int row = blockIdx.x, i = threadIdx.x;
  s[i] = nu[(size_t)row * BLOCK + i] * 2.f - 1.f;
  k[i] = impulse[(size_t)row * BLOCK + i];
  __syncthreads();
  float acc = 0.f;
  for (int m = 0; m <= i; ++m) acc += s[m] * k[i - m];
  out[(size_t)row * BLOCK + i] = acc;
}

__global__ void frame_phase(const float* pitch, float* fph, int n) {
  int i = blockIdx.x * blockDim.x + threadIdx.x;
  if (i < n) fph[i] = (float)BLOCK * TWO_PI * pitch[i] / (float)SRATE;
}

__global__ void frame_scan(const float* fph, float* foff, int T) {
  int b = blockIdx.x;
  if (threadIdx.x == 0) {
    float acc = 0.f;
    for (int t = 0; t < T; ++t) { foff[b * T + t] = acc; acc += fph[b * T + t]; }
  }
}

// harmonic additive synth + add filtered noise -> signal (B,64000)
__global__ __launch_bounds__(160) void harmonic_kernel(const float* __restrict__ pitch,
                                                       const float* __restrict__ foff,
                                                       const float* __restrict__ amps,
                                                       const float* __restrict__ noise,
                                                       float* __restrict__ signal) {
  __shared__ float a[NHARM];
  int row = blockIdx.x, i = threadIdx.x;
  for (int k = i; k < NHARM; k += 160) a[k] = amps[(size_t)row * NHARM + k];
  __syncthreads();
  float dphi = TWO_PI * pitch[row] / (float)SRATE;
  float om = foff[row] + dphi * (float)(i + 1);   // inclusive cumsum
  float acc = 0.f;
#pragma unroll 4
  for (int h = 1; h <= NHARM; ++h) acc += __sinf(om * (float)h) * a[h - 1];
  signal[(size_t)row * BLOCK + i] = acc + noise[(size_t)row * BLOCK + i];
}

__global__ void rev_imp_kernel(const float* __restrict__ rev_noise,
                               const float* decay, const float* wet,
                               float* __restrict__ imp) {
  int i = blockIdx.x * blockDim.x + threadIdx.x;
  if (i >= SRATE) return;
  float sp = log1pf(__expf(-decay[0]));             // softplus(-decay)
  float env = __expf(-sp * ((float)i / (float)SRATE) * 500.f);
  float w = 1.f / (1.f + __expf(-wet[0]));
  imp[i] = (i == 0) ? 1.f : rev_noise[i] * env * w;
}

// reverb: causal 16000-tap FIR, LDS-tiled. grid=(250,B), block=256
#define RT 256
__global__ __launch_bounds__(RT) void reverb_conv(const float* __restrict__ signal,
                                                  const float* __restrict__ imp,
                                                  float* __restrict__ out) {
  __shared__ float ksh[RT];
  __shared__ float ssh[2 * RT];
  int b = blockIdx.y;
  int o0 = blockIdx.x * RT;
  int tid = threadIdx.x;
  const float* s = signal + (size_t)b * SIGLEN;
  float acc = 0.f;
  for (int c = 0; c < SRATE; c += RT) {
    ksh[tid] = (c + tid < SRATE) ? imp[c + tid] : 0.f;
    int base = o0 - c - RT + 1;
    for (int l = tid; l < 2 * RT; l += RT) {
      int idx = base + l;
      ssh[l] = (idx >= 0 && idx < SIGLEN) ? s[idx] : 0.f;
    }
    __syncthreads();
#pragma unroll 8
    for (int p = 0; p < RT; ++p) acc += ssh[RT - 1 + tid - p] * ksh[p];
    __syncthreads();
  }
  out[(size_t)b * SIGLEN + o0 + tid] = acc;
}

// ---------------------------------------------------------------------------
// Host orchestration
// ---------------------------------------------------------------------------
static inline void cvt16(const float* s, f16* d, int n, hipStream_t st) {
  cvt_f16_kernel<<<(n + 255) / 256, 256, 0, st>>>(s, d, n);
}
static inline void gemm(const float* A, int lda, const f16* W, const float* bias,
                        float* C, int ldc, int M, int N, int K, hipStream_t st) {
  dim3 g((N + 127) / 128, (M + 63) / 64);
  gemm_bias_wmma<<<g, 256, 0, st>>>(A, lda, W, bias, C, ldc, M, N, K);
}
static inline void lnact(const float* s, int lds, float* d, int ldd,
                         const float* g, const float* b, int w, int leaky,
                         int rows, hipStream_t st) {
  ln_act<<<rows, 256, 0, st>>>(s, lds, d, ldd, g, b, w, leaky);
}

extern "C" void kernel_launch(void* const* d_in, const int* in_sizes, int n_in,
                              void* d_out, int out_size, void* d_ws, size_t ws_size,
                              hipStream_t stream) {
  // -------- input map (setup_inputs dict order, params flattened in order) ----
  const float* pitch   = (const float*)d_in[0];   // (8,400,1)
  const float* loud    = (const float*)d_in[1];
  const float* mfcc    = (const float*)d_in[2];   // (8,400,30)
  const float* eps     = (const float*)d_in[5];   // (8,400,16)
  const float* noise_u = (const float*)d_in[6];   // (8,400,160)
  int pi = 7;
  const float* enc_ln_g = (const float*)d_in[pi++];
  const float* enc_ln_b = (const float*)d_in[pi++];
  const float* encWih   = (const float*)d_in[pi++];   // (1536,30)
  const float* encWhh   = (const float*)d_in[pi++];   // (1536,512)
  const float* enc_bih  = (const float*)d_in[pi++];
  const float* enc_bhh  = (const float*)d_in[pi++];
  const float* fcmu_w   = (const float*)d_in[pi++];   // (16,512)
  const float* fcmu_b   = (const float*)d_in[pi++];
  const float* fclv_w   = (const float*)d_in[pi++];
  const float* fclv_b   = (const float*)d_in[pi++];
  const float* mlp01_w[3], *mlp01_b[3], *mlp01_g[3], *mlp01_e[3];
  for (int l = 0; l < 3; ++l) { mlp01_w[l]=(const float*)d_in[pi++]; mlp01_b[l]=(const float*)d_in[pi++];
                                mlp01_g[l]=(const float*)d_in[pi++]; mlp01_e[l]=(const float*)d_in[pi++]; }
  const float* mlp2_w[3], *mlp2_b[3], *mlp2_g[3], *mlp2_e[3];
  for (int l = 0; l < 3; ++l) { mlp2_w[l]=(const float*)d_in[pi++]; mlp2_b[l]=(const float*)d_in[pi++];
                                mlp2_g[l]=(const float*)d_in[pi++]; mlp2_e[l]=(const float*)d_in[pi++]; }
  const float* decWih   = (const float*)d_in[pi++];   // (1536,1536)
  const float* decWhh   = (const float*)d_in[pi++];   // (1536,512)
  const float* dec_bih  = (const float*)d_in[pi++];
  const float* dec_bhh  = (const float*)d_in[pi++];
  const float* out_w[3], *out_b[3], *out_g[3], *out_e[3];
  for (int l = 0; l < 3; ++l) { out_w[l]=(const float*)d_in[pi++]; out_b[l]=(const float*)d_in[pi++];
                                out_g[l]=(const float*)d_in[pi++]; out_e[l]=(const float*)d_in[pi++]; }
  const float* proj0_w = (const float*)d_in[pi++];    // (65,512)
  const float* proj0_b = (const float*)d_in[pi++];
  const float* proj1_w = (const float*)d_in[pi++];
  const float* proj1_b = (const float*)d_in[pi++];
  const float* rev_noise = (const float*)d_in[pi++];  // (16000,1)
  const float* rev_decay = (const float*)d_in[pi++];
  const float* rev_wet   = (const float*)d_in[pi++];

  // -------- workspace bump allocator ----------------------------------------
  size_t off = 0;
  auto alloc = [&](size_t bytes) -> void* {
    off = (off + 255) & ~(size_t)255;
    void* p = (char*)d_ws + off;
    off += bytes;
    return p;
  };
  auto f16buf = [&](size_t n) { return (f16*)alloc(n * sizeof(f16)); };
  auto f32buf = [&](size_t n) { return (float*)alloc(n * sizeof(float)); };

  // f16 weight copies (resident in L2 after first use)
  f16* encWih16 = f16buf(1536 * 30);
  f16* encWhh16 = f16buf(1536 * 512);
  f16* fcmu16   = f16buf(16 * 512);
  f16* fclv16   = f16buf(16 * 512);
  f16* m01w16[3] = { f16buf(512 * 1),  f16buf(512 * 512), f16buf(512 * 512) };
  f16* m2w16[3]  = { f16buf(512 * 16), f16buf(512 * 512), f16buf(512 * 512) };
  f16* decWih16 = f16buf(1536 * 1536);
  f16* decWhh16 = f16buf(1536 * 512);
  f16* outw16[3] = { f16buf(512 * 514), f16buf(512 * 512), f16buf(512 * 512) };
  f16* proj0_16 = f16buf(65 * 512);
  f16* proj1_16 = f16buf(65 * 512);

  // f32 activation buffers
  float* xln     = f32buf(ROWS * 30);
  float* xw_buf  = f32buf((size_t)ROWS * 1536);  // enc_xw then reused as dec_xw
  float* enc_hs  = f32buf((size_t)ROWS * HID);
  float* mu      = f32buf(ROWS * LAT);
  float* lv      = f32buf(ROWS * LAT);
  float* zz      = f32buf(ROWS * LAT);
  float* tmpA    = f32buf((size_t)ROWS * HID);
  float* tmpB    = f32buf((size_t)ROWS * HID);
  float* hidden  = f32buf((size_t)ROWS * 1536);
  float* dec_hs  = f32buf((size_t)ROWS * HID);
  float* hidden2 = f32buf((size_t)ROWS * 514);
  float* plin    = f32buf(ROWS * NBANDS);
  float* blin    = f32buf(ROWS * NBANDS);
  float* amps    = f32buf(ROWS * NHARM);
  float* impulse = f32buf((size_t)ROWS * BLOCK);
  float* noisef  = f32buf((size_t)ROWS * BLOCK);
  float* signal  = f32buf((size_t)ROWS * BLOCK);
  float* fph     = f32buf(ROWS);
  float* foff    = f32buf(ROWS);
  float* rimp    = f32buf(SRATE);
  (void)ws_size; (void)n_in; (void)in_sizes; (void)out_size;

  // -------- weight conversions ----------------------------------------------
  cvt16(encWih, encWih16, 1536 * 30, stream);
  cvt16(encWhh, encWhh16, 1536 * 512, stream);
  cvt16(fcmu_w, fcmu16, 16 * 512, stream);
  cvt16(fclv_w, fclv16, 16 * 512, stream);
  int m01K[3] = { 1, 512, 512 }, m2K[3] = { 16, 512, 512 }, outK[3] = { 514, 512, 512 };
  for (int l = 0; l < 3; ++l) {
    cvt16(mlp01_w[l], m01w16[l], 512 * m01K[l], stream);
    cvt16(mlp2_w[l],  m2w16[l],  512 * m2K[l],  stream);
    cvt16(out_w[l],   outw16[l], 512 * outK[l], stream);
  }
  cvt16(decWih, decWih16, 1536 * 1536, stream);
  cvt16(decWhh, decWhh16, 1536 * 512, stream);
  cvt16(proj0_w, proj0_16, 65 * 512, stream);
  cvt16(proj1_w, proj1_16, 65 * 512, stream);

  // -------- encoder ----------------------------------------------------------
  lnact(mfcc, 30, xln, 30, enc_ln_g, enc_ln_b, 30, 0, ROWS, stream);
  gemm(xln, 30, encWih16, enc_bih, xw_buf, 1536, ROWS, 1536, 30, stream);
  gru_seq<<<1, 1024, 0, stream>>>(xw_buf, encWhh16, enc_bhh, enc_hs, TT);
  gemm(enc_hs, HID, fcmu16, fcmu_b, mu, LAT, ROWS, LAT, HID, stream);
  gemm(enc_hs, HID, fclv16, fclv_b, lv, LAT, ROWS, LAT, HID, stream);
  z_kernel<<<(ROWS * LAT + 255) / 256, 256, 0, stream>>>(mu, lv, eps, zz, ROWS * LAT);

  // -------- three MLPs -> hidden (ROWS x 1536) -------------------------------
  auto run_mlp = [&](const float* in0, int K0, f16* const* w16, const float* const* b,
                     const float* const* g, const float* const* e, float* slice) {
    gemm(in0, K0, w16[0], b[0], tmpA, HID, ROWS, HID, K0, stream);
    lnact(tmpA, HID, tmpA, HID, g[0], e[0], HID, 1, ROWS, stream);
    gemm(tmpA, HID, w16[1], b[1], tmpB, HID, ROWS, HID, HID, stream);
    lnact(tmpB, HID, tmpB, HID, g[1], e[1], HID, 1, ROWS, stream);
    gemm(tmpB, HID, w16[2], b[2], slice, 1536, ROWS, HID, HID, stream);
    lnact(slice, 1536, slice, 1536, g[2], e[2], HID, 1, ROWS, stream);
  };
  run_mlp(pitch, 1,  m01w16, mlp01_b, mlp01_g, mlp01_e, hidden + 0);
  run_mlp(loud,  1,  m01w16, mlp01_b, mlp01_g, mlp01_e, hidden + HID);
  run_mlp(zz,    16, m2w16,  mlp2_b,  mlp2_g,  mlp2_e,  hidden + 2 * HID);

  // -------- decoder GRU ------------------------------------------------------
  gemm(hidden, 1536, decWih16, dec_bih, xw_buf, 1536, ROWS, 1536, 1536, stream);
  gru_seq<<<1, 1024, 0, stream>>>(xw_buf, decWhh16, dec_bhh, dec_hs, TT);
  concat_hidden2<<<(ROWS * 514 + 255) / 256, 256, 0, stream>>>(dec_hs, pitch, loud,
                                                               hidden2, ROWS * 514);

  // -------- output MLP + projections -----------------------------------------
  gemm(hidden2, 514, outw16[0], out_b[0], tmpA, HID, ROWS, HID, 514, stream);
  lnact(tmpA, HID, tmpA, HID, out_g[0], out_e[0], HID, 1, ROWS, stream);
  gemm(tmpA, HID, outw16[1], out_b[1], tmpB, HID, ROWS, HID, HID, stream);
  lnact(tmpB, HID, tmpB, HID, out_g[1], out_e[1], HID, 1, ROWS, stream);
  gemm(tmpB, HID, outw16[2], out_b[2], tmpA, HID, ROWS, HID, HID, stream);
  lnact(tmpA, HID, tmpA, HID, out_g[2], out_e[2], HID, 1, ROWS, stream);
  gemm(tmpA, HID, proj0_16, proj0_b, plin, NBANDS, ROWS, NBANDS, HID, stream);
  gemm(tmpA, HID, proj1_16, proj1_b, blin, NBANDS, ROWS, NBANDS, HID, stream);

  // -------- synthesis --------------------------------------------------------
  amps_kernel<<<ROWS, 64, 0, stream>>>(plin, pitch, amps);
  impulse_kernel<<<ROWS, 160, 0, stream>>>(blin, impulse);
  noise_conv<<<ROWS, 160, 0, stream>>>(noise_u, impulse, noisef);
  frame_phase<<<(ROWS + 255) / 256, 256, 0, stream>>>(pitch, fph, ROWS);
  frame_scan<<<BB, 32, 0, stream>>>(fph, foff, TT);
  harmonic_kernel<<<ROWS, 160, 0, stream>>>(pitch, foff, amps, noisef, signal);

  // -------- reverb -----------------------------------------------------------
  rev_imp_kernel<<<(SRATE + 255) / 256, 256, 0, stream>>>(rev_noise, rev_decay,
                                                          rev_wet, rimp);
  dim3 rg(SIGLEN / RT, BB);
  reverb_conv<<<rg, RT, 0, stream>>>(signal, rimp, (float*)d_out);
}
